// GATNet_28303834481266
// MI455X (gfx1250) — matrix operand order repaired and checked by reading, not compiled, not run on hardware
//
#include <hip/hip_runtime.h>
#include <math.h>

// ---------- CDNA5 WMMA types ----------
typedef __attribute__((ext_vector_type(16))) __bf16        v16bf;
typedef __attribute__((ext_vector_type(8)))  __bf16        v8bf;
typedef __attribute__((ext_vector_type(4)))  __bf16        v4bf;
typedef __attribute__((ext_vector_type(8)))  float         v8f;
typedef __attribute__((ext_vector_type(4)))  unsigned int  v4u;

#define HC   64      // layer-1 output width (8 heads x 8 ch) == W1 cols
#define TM   32      // rows per block tile
#define TK   32      // K chunk (== WMMA K for bf16)
#define ASTR 40      // LDS row stride (bf16 units): 80B, 16B-aligned, conflict-reducing

__device__ __forceinline__ void atomicMaxF(float* a, float v) {
  // sign-aware monotone bit trick; init value must be -inf (0xFF800000)
  if (v >= 0.f) atomicMax((int*)a, __float_as_int(v));
  else          atomicMin((unsigned int*)a, __float_as_uint(v));
}

__device__ __forceinline__ float lrelu(float v) { return v > 0.f ? v : 0.2f * v; }

// ================= Layer-1 fused gather + bf16 WMMA GEMM =================
// hs[m, 0:64] = x[n_id[m], 0:K] @ W[0:K, 0:64]
__global__ __launch_bounds__(256) void k_gemm1_wmma(
    const float* __restrict__ x, const int* __restrict__ n_id,
    const float* __restrict__ W, float* __restrict__ hs, int Nrows, int K) {
  __shared__ __align__(16) __bf16 As[TM * ASTR];  // [row][k]
  __shared__ __align__(16) __bf16 Bs[HC * ASTR];  // [col][k]  (transposed W chunk)

  const int t    = threadIdx.x;
  const int lane = t & 31;
  const int wave = t >> 5;        // 0..7
  const int wm   = wave >> 2;     // 0..1 : 16-row sub-tile
  const int wn   = wave & 3;      // 0..3 : 16-col sub-tile
  const int row0 = blockIdx.x * TM;
  const int half = lane >> 4;     // K-half selector per ISA layout
  const int ml   = lane & 15;

  // staging coordinates (fixed per thread)
  const int ar = t >> 3;          // A row 0..31
  const int ak = (t & 7) * 4;     // A k offset (4 consecutive)
  const int bc = t & 63;          // B column 0..63
  const int bk = (t >> 6) * 8;    // B k offset (8 consecutive)

  int grow = row0 + ar; grow = grow < Nrows ? grow : Nrows - 1;
  const float* __restrict__ xrow = x + (long)n_id[grow] * K;  // gathered row base
  const float* __restrict__ wcol = W + bc;                    // W column, stride HC

  // fragment LDS base pointers (fixed per lane)
  const __bf16* Ap = &As[(wm * 16 + ml) * ASTR];
  const __bf16* Bp = &Bs[(wn * 16 + ml) * ASTR + 8 * half];

  v8f acc = {};
  const int kfull = (K / TK) * TK;

  // ---------------- main loop: no bounds checks ----------------
  for (int k0 = 0; k0 < kfull; k0 += TK) {
    // A: 4 consecutive floats (x rows only 8B-aligned -> two b64 loads)
    float2 xa0 = *(const float2*)(xrow + k0 + ak);
    float2 xa1 = *(const float2*)(xrow + k0 + ak + 2);
    v4bf pa = { (__bf16)xa0.x, (__bf16)xa0.y, (__bf16)xa1.x, (__bf16)xa1.y };
    *(v4bf*)(&As[ar * ASTR + ak]) = pa;                       // ds_store_b64
    // B: 8 strided (coalesced-across-lanes) floats -> one b128 LDS store
    v8bf pb;
#pragma unroll
    for (int j = 0; j < 8; ++j) pb[j] = (__bf16)wcol[(long)(k0 + bk + j) * HC];
    *(v8bf*)(&Bs[bc * ASTR + bk]) = pb;                       // ds_store_b128
    __syncthreads();

    union Frag { v16bf v; v4u q[2]; } a, b;
    a.q[0] = *(const v4u*)(Ap + 8 * half);        // elems 0..7  : K = 8*half+0..7
    a.q[1] = *(const v4u*)(Ap + 16 + 8 * half);   // elems 8..15 : K = 16+8*half+0..7
    b.q[0] = *(const v4u*)(Bp);                   // elems 0..7  : K = 8*half+0..7
    b.q[1] = *(const v4u*)(Bp + 16);              // elems 8..15 : K = 16+8*half+0..7
    acc = __builtin_amdgcn_wmma_f32_16x16x32_bf16(false, a.v, false, b.v,
                                                  (short)0, acc, false, false);
    __syncthreads();
  }

  // ---------------- tail chunk: zero-padded ----------------
  if (kfull < K) {
    const int k0 = kfull;
    v4bf pa;
#pragma unroll
    for (int j = 0; j < 4; ++j) {
      int k = k0 + ak + j;
      pa[j] = (__bf16)((k < K) ? xrow[k] : 0.f);
    }
    *(v4bf*)(&As[ar * ASTR + ak]) = pa;
    v8bf pb;
#pragma unroll
    for (int j = 0; j < 8; ++j) {
      int k = k0 + bk + j;
      pb[j] = (__bf16)((k < K) ? wcol[(long)k * HC] : 0.f);
    }
    *(v8bf*)(&Bs[bc * ASTR + bk]) = pb;
    __syncthreads();

    union Frag { v16bf v; v4u q[2]; } a, b;
    a.q[0] = *(const v4u*)(Ap + 8 * half);
    a.q[1] = *(const v4u*)(Ap + 16 + 8 * half);
    b.q[0] = *(const v4u*)(Bp);
    b.q[1] = *(const v4u*)(Bp + 16);
    acc = __builtin_amdgcn_wmma_f32_16x16x32_bf16(false, a.v, false, b.v,
                                                  (short)0, acc, false, false);
    __syncthreads();
  }

  // ---- store C/D: VGPR v -> M = wm*16 + 8*half + v, N = wn*16 + ml ----
  const int col = wn * 16 + ml;
#pragma unroll
  for (int v = 0; v < 8; ++v) {
    int m = row0 + wm * 16 + 8 * half + v;
    if (m < Nrows) hs[(long)m * HC + col] = acc[v];
  }
}

// ================= small helper kernels =================
__global__ void k_fill(float* p, float v, long n) {
  long i = (long)blockIdx.x * blockDim.x + threadIdx.x;
  if (i < n) p[i] = v;
}

// s[n,h] = dot(hs1[n, h*8 : h*8+8], att[h, :])     (H=8, C=8)
__global__ void k_dot8(const float* __restrict__ hs, const int* __restrict__ gather,
                       const float* __restrict__ att, float* __restrict__ s, int n) {
  int i = blockIdx.x * blockDim.x + threadIdx.x;
  if (i >= n * 8) return;
  int r = i >> 3, h = i & 7;
  long node = gather ? (long)gather[r] : (long)r;
  const float* hr = hs + node * HC + h * 8;
  const float* ar = att + h * 8;
  float acc = 0.f;
#pragma unroll
  for (int c = 0; c < 8; ++c) acc += hr[c] * ar[c];
  s[i] = acc;
}

__global__ void k_edge_max1(const int* src, const int* dst, const float* ssrc,
                            const float* sdst, float* m, int E) {
  int i = blockIdx.x * blockDim.x + threadIdx.x;
  if (i >= E * 8) return;
  int e = i >> 3, h = i & 7;
  float v = lrelu(ssrc[src[e] * 8 + h] + sdst[dst[e] * 8 + h]);
  atomicMaxF(&m[dst[e] * 8 + h], v);
}

__global__ void k_edge_sum1(const int* src, const int* dst, const float* ssrc,
                            const float* sdst, const float* m, float* denom, int E) {
  int i = blockIdx.x * blockDim.x + threadIdx.x;
  if (i >= E * 8) return;
  int e = i >> 3, h = i & 7;
  int d = dst[e];
  float v = lrelu(ssrc[src[e] * 8 + h] + sdst[d * 8 + h]);
  atomicAdd(&denom[d * 8 + h], expf(v - m[d * 8 + h]));
}

__global__ void k_edge_agg1(const int* src, const int* dst, const float* ssrc,
                            const float* sdst, const float* m, const float* denom,
                            const float* __restrict__ hs, float* agg, int E) {
  int i = blockIdx.x * blockDim.x + threadIdx.x;
  if (i >= E * 8) return;
  int e = i >> 3, h = i & 7;
  int d = dst[e], sN = src[e];
  float v = lrelu(ssrc[sN * 8 + h] + sdst[d * 8 + h]);
  float coef = expf(v - m[d * 8 + h]) / denom[d * 8 + h];
  const float* hr = hs + (long)sN * HC + h * 8;
  float* ar = agg + (long)d * HC + h * 8;
#pragma unroll
  for (int c = 0; c < 8; ++c) atomicAdd(&ar[c], coef * hr[c]);
}

// out1 = elu(agg + bias)  (in place on agg)
__global__ void k_finalize1(float* agg, const float* bias, int n) {
  int i = blockIdx.x * blockDim.x + threadIdx.x;
  if (i >= n) return;
  float v = agg[i] + bias[i & (HC - 1)];
  agg[i] = v > 0.f ? v : expf(v) - 1.f;
}

// hs2[n, c] = dot(h[n, 0:64], W2[0:64, c])
__global__ void k_gemm2(const float* __restrict__ h, const float* __restrict__ W2,
                        float* hs2, int N, int OUT) {
  int i = blockIdx.x * blockDim.x + threadIdx.x;
  if (i >= N * OUT) return;
  int n = i / OUT, c = i % OUT;
  const float* hr = h + (long)n * HC;
  float acc = 0.f;
#pragma unroll 8
  for (int k = 0; k < HC; ++k) acc += hr[k] * W2[k * OUT + c];
  hs2[i] = acc;
}

// s[n] = dot(hs2[gather?gather[n]:n, 0:OUT], att[0:OUT])
__global__ void k_dot_vec(const float* __restrict__ hs2, const int* __restrict__ gather,
                          const float* __restrict__ att, float* s, int n, int OUT) {
  int i = blockIdx.x * blockDim.x + threadIdx.x;
  if (i >= n) return;
  long node = gather ? (long)gather[i] : (long)i;
  const float* hr = hs2 + node * OUT;
  float acc = 0.f;
  for (int c = 0; c < OUT; ++c) acc += hr[c] * att[c];
  s[i] = acc;
}

__global__ void k_edge_max2(const int* src, const int* dst, const float* ssrc,
                            const float* sdst, float* m, int E) {
  int e = blockIdx.x * blockDim.x + threadIdx.x;
  if (e >= E) return;
  atomicMaxF(&m[dst[e]], lrelu(ssrc[src[e]] + sdst[dst[e]]));
}

__global__ void k_edge_sum2(const int* src, const int* dst, const float* ssrc,
                            const float* sdst, const float* m, float* denom, int E) {
  int e = blockIdx.x * blockDim.x + threadIdx.x;
  if (e >= E) return;
  int d = dst[e];
  atomicAdd(&denom[d], expf(lrelu(ssrc[src[e]] + sdst[d]) - m[d]));
}

__global__ void k_edge_agg2(const int* src, const int* dst, const float* ssrc,
                            const float* sdst, const float* m, const float* denom,
                            const float* __restrict__ hs2, float* agg, int E, int OUT) {
  int i = blockIdx.x * blockDim.x + threadIdx.x;
  if (i >= E * OUT) return;
  int e = i / OUT, c = i % OUT;
  int d = dst[e], sN = src[e];
  float coef = expf(lrelu(ssrc[sN] + sdst[d]) - m[d]) / denom[d];
  atomicAdd(&agg[(long)d * OUT + c], coef * hs2[(long)sN * OUT + c]);
}

__global__ void k_logsoftmax(const float* __restrict__ agg, const float* __restrict__ bias,
                             float* out, int Nd, int OUT) {
  int d = blockIdx.x * blockDim.x + threadIdx.x;
  if (d >= Nd) return;
  const float* r = agg + (long)d * OUT;
  float mx = -INFINITY;
  for (int c = 0; c < OUT; ++c) mx = fmaxf(mx, r[c] + bias[c]);
  float s = 0.f;
  for (int c = 0; c < OUT; ++c) s += expf(r[c] + bias[c] - mx);
  float lse = mx + logf(s);
  for (int c = 0; c < OUT; ++c) out[(long)d * OUT + c] = r[c] + bias[c] - lse;
}

// ================= host-side launch =================
static inline int cdiv(long a, int b) { return (int)((a + b - 1) / b); }

extern "C" void kernel_launch(void* const* d_in, const int* in_sizes, int n_in,
                              void* d_out, int out_size, void* d_ws, size_t ws_size,
                              hipStream_t stream) {
  const float* x        = (const float*)d_in[0];
  const int*   n_id1    = (const int*)d_in[1];
  const int*   res1     = (const int*)d_in[2];
  const int*   esrc1    = (const int*)d_in[3];
  const int*   edst1    = (const int*)d_in[4];
  const int*   res2     = (const int*)d_in[5];
  const int*   esrc2    = (const int*)d_in[6];
  const int*   edst2    = (const int*)d_in[7];
  const float* W1       = (const float*)d_in[8];
  const float* a_src1   = (const float*)d_in[9];
  const float* a_dst1   = (const float*)d_in[10];
  const float* bias1    = (const float*)d_in[11];
  const float* W2       = (const float*)d_in[12];
  const float* a_src2   = (const float*)d_in[13];
  const float* a_dst2   = (const float*)d_in[14];
  const float* bias2    = (const float*)d_in[15];

  const int N1  = in_sizes[1];          // 120000
  const int N1D = in_sizes[2];          // 30000
  const int E1  = in_sizes[3];          // 960000
  const int N2D = in_sizes[5];          // 6000
  const int E2  = in_sizes[6];          // 192000
  const int OUT = in_sizes[15];         // 41
  const int K   = in_sizes[8] / HC;     // 602

  // ---- workspace carve (floats); total ~51 MB -> fits in 192MB L2 ----
  float* ws      = (float*)d_ws;
  float* hs1     = ws; ws += (long)N1 * HC;    // [N1, 64]
  float* ssrc1   = ws; ws += (long)N1 * 8;     // [N1, 8]
  float* sdst1   = ws; ws += (long)N1D * 8;
  float* m1      = ws; ws += (long)N1D * 8;
  float* denom1  = ws; ws += (long)N1D * 8;
  float* agg1    = ws; ws += (long)N1D * HC;   // becomes out1 (elu) in place
  float* hs2     = ws; ws += (long)N1D * OUT;
  float* ssrc2   = ws; ws += (long)N1D;
  float* sdst2   = ws; ws += (long)N2D;
  float* m2      = ws; ws += (long)N2D;
  float* denom2  = ws; ws += (long)N2D;
  float* agg2    = ws; ws += (long)N2D * OUT;

  const int B = 256;

  // Layer 1: fused gather + WMMA GEMM
  k_gemm1_wmma<<<cdiv(N1, TM), B, 0, stream>>>(x, n_id1, W1, hs1, N1, K);

  // attention projections
  k_dot8<<<cdiv((long)N1 * 8, B), B, 0, stream>>>(hs1, nullptr, a_src1, ssrc1, N1);
  k_dot8<<<cdiv((long)N1D * 8, B), B, 0, stream>>>(hs1, res1, a_dst1, sdst1, N1D);

  // init segment buffers
  k_fill<<<cdiv((long)N1D * 8, B), B, 0, stream>>>(m1, -INFINITY, (long)N1D * 8);
  k_fill<<<cdiv((long)N1D * 8, B), B, 0, stream>>>(denom1, 0.f, (long)N1D * 8);
  k_fill<<<cdiv((long)N1D * HC, B), B, 0, stream>>>(agg1, 0.f, (long)N1D * HC);

  // segment softmax + weighted scatter aggregation (3 passes, logits recomputed)
  k_edge_max1<<<cdiv((long)E1 * 8, B), B, 0, stream>>>(esrc1, edst1, ssrc1, sdst1, m1, E1);
  k_edge_sum1<<<cdiv((long)E1 * 8, B), B, 0, stream>>>(esrc1, edst1, ssrc1, sdst1, m1, denom1, E1);
  k_edge_agg1<<<cdiv((long)E1 * 8, B), B, 0, stream>>>(esrc1, edst1, ssrc1, sdst1, m1, denom1,
                                                       hs1, agg1, E1);
  k_finalize1<<<cdiv((long)N1D * HC, B), B, 0, stream>>>(agg1, bias1, N1D * HC);

  // Layer 2
  k_gemm2<<<cdiv((long)N1D * OUT, B), B, 0, stream>>>(agg1, W2, hs2, N1D, OUT);
  k_dot_vec<<<cdiv(N1D, B), B, 0, stream>>>(hs2, nullptr, a_src2, ssrc2, N1D, OUT);
  k_dot_vec<<<cdiv(N2D, B), B, 0, stream>>>(hs2, res2, a_dst2, sdst2, N2D, OUT);

  k_fill<<<cdiv(N2D, B), B, 0, stream>>>(m2, -INFINITY, N2D);
  k_fill<<<cdiv(N2D, B), B, 0, stream>>>(denom2, 0.f, N2D);
  k_fill<<<cdiv((long)N2D * OUT, B), B, 0, stream>>>(agg2, 0.f, (long)N2D * OUT);

  k_edge_max2<<<cdiv(E2, B), B, 0, stream>>>(esrc2, edst2, ssrc2, sdst2, m2, E2);
  k_edge_sum2<<<cdiv(E2, B), B, 0, stream>>>(esrc2, edst2, ssrc2, sdst2, m2, denom2, E2);
  k_edge_agg2<<<cdiv((long)E2 * OUT, B), B, 0, stream>>>(esrc2, edst2, ssrc2, sdst2, m2, denom2,
                                                         hs2, agg2, E2, OUT);

  k_logsoftmax<<<cdiv(N2D, B), B, 0, stream>>>(agg2, bias2, (float*)d_out, N2D, OUT);
}
